// NonLocal_27127013442144
// MI455X (gfx1250) — compile-verified, matrix operand order checked by reading, not run on hardware
//
#include <hip/hip_runtime.h>

// Non-local block, MI455X (gfx1250), wave32 + V_WMMA_F32_16X16X32_BF16.
// b=16, c=1024, n=2304, ci=256. Flash-fused attention (no f materialization).

#define N_POS 2304
#define CIN   1024
#define CI    256
#define BATCH 16

typedef __attribute__((ext_vector_type(16))) __bf16 v16bf;
typedef __attribute__((ext_vector_type(8)))  float  v8f;

union FragB { v16bf v; uint4 q[2]; };

// Load a 16x32 bf16 A (or B^T) fragment from a row-major matrix (leading dim ld).
// Per ISA 16-bit A layout: lane L holds row L%16; K chunk (L/16)*8..+7 in VGPR0-3,
// +16..+23 in VGPR4-7  ->  two contiguous 16B loads per lane.
__device__ __forceinline__ v16bf load_frag(const __bf16* __restrict__ base, int ld, int lane) {
    const int row = lane & 15;
    const int kc  = (lane >> 4) << 3;
    FragB f;
    f.q[0] = *(const uint4*)(base + (size_t)row * ld + kc);
    f.q[1] = *(const uint4*)(base + (size_t)row * ld + kc + 16);
    return f.v;
}

__device__ __forceinline__ v8f wmma_bf16(v16bf a, v16bf b, v8f c) {
    return __builtin_amdgcn_wmma_f32_16x16x32_bf16(false, a, false, b, (short)0, c, false, false);
}

__device__ __forceinline__ int c_row(int r, int lane) { return r + ((lane >> 4) << 3); }

// ---------------- weights fp32 -> bf16 (layouts preserved) ----------------
__global__ void k_cvt_weights(const float* __restrict__ gw, const float* __restrict__ tw,
                              const float* __restrict__ pw, const float* __restrict__ Ww,
                              __bf16* __restrict__ out) {
    const int SZ = CI * CIN;                       // 262144 each
    int i = blockIdx.x * blockDim.x + threadIdx.x;
    if (i >= 4 * SZ) return;
    int sel = i / SZ, j = i - sel * SZ;
    float v = (sel == 0) ? gw[j] : (sel == 1) ? tw[j] : (sel == 2) ? pw[j] : Ww[j];
    out[i] = (__bf16)v;
}

// ---------------- x (b,c,n) f32 -> xT (b,n,c) bf16, LDS tiled ----------------
__global__ void k_transpose_x(const float* __restrict__ x, __bf16* __restrict__ xT) {
    __shared__ float tile[32][33];
    const int b = blockIdx.z;
    const int n0 = blockIdx.x * 32, c0 = blockIdx.y * 32;
    const int tx = threadIdx.x, ty = threadIdx.y;
    const float* xb = x + (size_t)b * CIN * N_POS;
#pragma unroll
    for (int i = 0; i < 4; ++i)
        tile[ty + 8 * i][tx] = xb[(size_t)(c0 + ty + 8 * i) * N_POS + n0 + tx];
    __syncthreads();
    __bf16* xTb = xT + (size_t)b * N_POS * CIN;
#pragma unroll
    for (int i = 0; i < 4; ++i)
        xTb[(size_t)(n0 + ty + 8 * i) * CIN + c0 + tx] = (__bf16)tile[tx][ty + 8 * i];
}

// ---------- theta/phi projections: (n,1024)@(1024,256) -> (n,256) bf16 ----------
// D[M=n][N=o]; A = xT rows, B^T = weight rows (o,c) as stored. grid(144,16), block 32.
__global__ void __launch_bounds__(32)
k_proj_tp(const __bf16* __restrict__ xT, const __bf16* __restrict__ tw,
          const __bf16* __restrict__ pw, const float* __restrict__ tb,
          const float* __restrict__ pb, __bf16* __restrict__ theta,
          __bf16* __restrict__ phi) {
    const int lane = threadIdx.x;
    const int nt = blockIdx.x, b = blockIdx.y;
    const __bf16* xr = xT + ((size_t)b * N_POS + nt * 16) * CIN;
    const v8f zero = {0, 0, 0, 0, 0, 0, 0, 0};
#pragma unroll 1
    for (int ot = 0; ot < CI / 16; ++ot) {
        v8f accT = zero, accP = zero;
        for (int k = 0; k < CIN; k += 32) {
            v16bf a  = load_frag(xr + k, CIN, lane);
            v16bf bt = load_frag(tw + (size_t)(ot * 16) * CIN + k, CIN, lane);
            v16bf bp = load_frag(pw + (size_t)(ot * 16) * CIN + k, CIN, lane);
            accT = wmma_bf16(a, bt, accT);
            accP = wmma_bf16(a, bp, accP);
        }
        const int o = ot * 16 + (lane & 15);
        const float tbi = tb[o], pbi = pb[o];
#pragma unroll
        for (int r = 0; r < 8; ++r) {
            size_t idx = ((size_t)b * N_POS + nt * 16 + c_row(r, lane)) * CI + o;
            theta[idx] = (__bf16)(accT[r] + tbi);
            phi[idx]   = (__bf16)(accP[r] + pbi);
        }
    }
}

// ---------- g projection, stored TRANSPOSED: gT (b, ci, n) bf16 ----------
// D[M=o][N=n]; A = g_w rows, B^T = xT rows. grid(144,16,16), block 32.
__global__ void __launch_bounds__(32)
k_proj_g(const __bf16* __restrict__ xT, const __bf16* __restrict__ gw,
         const float* __restrict__ gb, __bf16* __restrict__ gT) {
    const int lane = threadIdx.x;
    const int nt = blockIdx.x, ob = blockIdx.y, b = blockIdx.z;
    v8f acc = {0, 0, 0, 0, 0, 0, 0, 0};
    for (int k = 0; k < CIN; k += 32) {
        v16bf a  = load_frag(gw + (size_t)(ob * 16) * CIN + k, CIN, lane);
        v16bf bb = load_frag(xT + ((size_t)b * N_POS + nt * 16) * CIN + k, CIN, lane);
        acc = wmma_bf16(a, bb, acc);
    }
#pragma unroll
    for (int r = 0; r < 8; ++r) {
        int o = ob * 16 + c_row(r, lane);
        gT[((size_t)b * CI + o) * N_POS + nt * 16 + (lane & 15)] = (__bf16)(acc[r] + gb[o]);
    }
}

// ---------- flash attention: y = softmax(theta.phiT) @ g, per 16-row block ----------
// grid(144,16), block 32 (one wave). O accumulators: 16 tiles x v8f = 128 VGPRs.
__global__ void __launch_bounds__(32)
k_attention(const __bf16* __restrict__ theta, const __bf16* __restrict__ phi,
            const __bf16* __restrict__ gT, __bf16* __restrict__ y) {
    const int lane = threadIdx.x;
    const int nt = blockIdx.x, b = blockIdx.y;
    __shared__ __bf16 sP[16][48];   // 16x32 P tile, padded (row pitch 96B = 6*16B)

    const __bf16* th = theta + ((size_t)b * N_POS + nt * 16) * CI;
    const __bf16* ph = phi + (size_t)b * N_POS * CI;
    const __bf16* g  = gT + (size_t)b * CI * N_POS;

    const v8f zero = {0, 0, 0, 0, 0, 0, 0, 0};
    float mrun[8], lrun[8];
#pragma unroll
    for (int r = 0; r < 8; ++r) { mrun[r] = -1e30f; lrun[r] = 0.f; }
    v8f O[16];
#pragma unroll
    for (int t = 0; t < 16; ++t) O[t] = zero;

#pragma unroll 1
    for (int m0 = 0; m0 < N_POS; m0 += 32) {
        // S = theta(16x256) . phiT(256x32): two 16x16 C tiles, K-depth 256
        v8f s0 = zero, s1 = zero;
        for (int k = 0; k < CI; k += 32) {
            v16bf a  = load_frag(th + k, CI, lane);
            v16bf b0 = load_frag(ph + (size_t)m0 * CI + k, CI, lane);
            v16bf b1 = load_frag(ph + (size_t)(m0 + 16) * CI + k, CI, lane);
            s0 = wmma_bf16(a, b0, s0);
            s1 = wmma_bf16(a, b1, s1);
        }
        // online softmax; row M = r + 8*(lane>=16), N = lane%16 -> reduce over 16-lane halves
        float scl[8];
#pragma unroll
        for (int r = 0; r < 8; ++r) {
            float v = fmaxf(s0[r], s1[r]);
            v = fmaxf(v, __shfl_xor(v, 1));
            v = fmaxf(v, __shfl_xor(v, 2));
            v = fmaxf(v, __shfl_xor(v, 4));
            v = fmaxf(v, __shfl_xor(v, 8));
            float nm = fmaxf(mrun[r], v);
            float p0 = __expf(s0[r] - nm);
            float p1 = __expf(s1[r] - nm);
            float ps = p0 + p1;
            ps += __shfl_xor(ps, 1);
            ps += __shfl_xor(ps, 2);
            ps += __shfl_xor(ps, 4);
            ps += __shfl_xor(ps, 8);
            scl[r] = __expf(mrun[r] - nm);
            lrun[r] = lrun[r] * scl[r] + ps;
            mrun[r] = nm;
            int row = c_row(r, lane);
            sP[row][lane & 15]        = (__bf16)p0;   // C-layout -> LDS row-major
            sP[row][(lane & 15) + 16] = (__bf16)p1;
        }
#pragma unroll
        for (int t = 0; t < 16; ++t)
#pragma unroll
            for (int r = 0; r < 8; ++r) O[t][r] *= scl[r];
        __syncthreads();
        // P as A fragment (LDS ds_load_b128), accumulate O += P(16x32) . g(32x256)
        {
            const int row = lane & 15;
            const int kc  = (lane >> 4) << 3;
            FragB f;
            f.q[0] = *(const uint4*)(&sP[row][kc]);
            f.q[1] = *(const uint4*)(&sP[row][kc + 16]);
            v16bf pa = f.v;
#pragma unroll
            for (int t = 0; t < 16; ++t) {
                v16bf bg = load_frag(g + (size_t)(t * 16) * N_POS + m0, N_POS, lane);
                O[t] = wmma_bf16(pa, bg, O[t]);
            }
        }
        __syncthreads();
    }
    // normalize and store y (b, n, ci) bf16
#pragma unroll
    for (int r = 0; r < 8; ++r) {
        float inv = 1.f / lrun[r];
        size_t base = ((size_t)b * N_POS + nt * 16 + c_row(r, lane)) * CI;
#pragma unroll
        for (int t = 0; t < 16; ++t)
            y[base + t * 16 + (lane & 15)] = (__bf16)(O[t][r] * inv);
    }
}

// ---------- W projection + bias + residual: out(b,c,n) f32 ----------
// D[M=c][N=n]; A = W_w rows (1024,256), B^T = y rows (n,256). grid(144,64,16), block 32.
__global__ void __launch_bounds__(32)
k_out(const __bf16* __restrict__ y, const __bf16* __restrict__ Ww,
      const float* __restrict__ Wb, const float* __restrict__ x,
      float* __restrict__ out) {
    const int lane = threadIdx.x;
    const int nt = blockIdx.x, cb = blockIdx.y, b = blockIdx.z;
    v8f acc = {0, 0, 0, 0, 0, 0, 0, 0};
    for (int k = 0; k < CI; k += 32) {
        v16bf a  = load_frag(Ww + (size_t)(cb * 16) * CI + k, CI, lane);
        v16bf bb = load_frag(y + ((size_t)b * N_POS + nt * 16) * CI + k, CI, lane);
        acc = wmma_bf16(a, bb, acc);
    }
#pragma unroll
    for (int r = 0; r < 8; ++r) {
        int c = cb * 16 + c_row(r, lane);
        size_t idx = ((size_t)b * CIN + c) * N_POS + nt * 16 + (lane & 15);
        out[idx] = acc[r] + Wb[c] + x[idx];   // coalesced over n
    }
}

extern "C" void kernel_launch(void* const* d_in, const int* in_sizes, int n_in,
                              void* d_out, int out_size, void* d_ws, size_t ws_size,
                              hipStream_t stream) {
    const float* x   = (const float*)d_in[0];
    const float* g_w = (const float*)d_in[1];
    const float* g_b = (const float*)d_in[2];
    const float* t_w = (const float*)d_in[3];
    const float* t_b = (const float*)d_in[4];
    const float* p_w = (const float*)d_in[5];
    const float* p_b = (const float*)d_in[6];
    const float* W_w = (const float*)d_in[7];
    const float* W_b = (const float*)d_in[8];
    float* out = (float*)d_out;

    char* ws = (char*)d_ws;
    const size_t SZ_XT = (size_t)BATCH * N_POS * CIN * 2;   // 75497472
    const size_t SZ_P  = (size_t)BATCH * N_POS * CI * 2;    // 18874368
    __bf16* xT    = (__bf16*)(ws);
    __bf16* theta = (__bf16*)(ws + SZ_XT);
    __bf16* phi   = (__bf16*)(ws + SZ_XT + SZ_P);
    __bf16* gT    = (__bf16*)(ws + SZ_XT + 2 * SZ_P);
    __bf16* ybuf  = (__bf16*)(ws + SZ_XT + 3 * SZ_P);
    __bf16* wbf   = (__bf16*)(ws + SZ_XT + 4 * SZ_P);
    const size_t WSZ = (size_t)CI * CIN;                    // 262144 elements
    __bf16* gw_bf = wbf;
    __bf16* tw_bf = wbf + WSZ;
    __bf16* pw_bf = wbf + 2 * WSZ;
    __bf16* Ww_bf = wbf + 3 * WSZ;

    k_cvt_weights<<<dim3((4 * WSZ + 255) / 256), dim3(256), 0, stream>>>(
        g_w, t_w, p_w, W_w, wbf);
    k_transpose_x<<<dim3(N_POS / 32, CIN / 32, BATCH), dim3(32, 8), 0, stream>>>(x, xT);
    k_proj_tp<<<dim3(N_POS / 16, BATCH), dim3(32), 0, stream>>>(
        xT, tw_bf, pw_bf, t_b, p_b, theta, phi);
    k_proj_g<<<dim3(N_POS / 16, CI / 16, BATCH), dim3(32), 0, stream>>>(
        xT, gw_bf, g_b, gT);
    k_attention<<<dim3(N_POS / 16, BATCH), dim3(32), 0, stream>>>(theta, phi, gT, ybuf);
    k_out<<<dim3(N_POS / 16, CIN / 16, BATCH), dim3(32), 0, stream>>>(
        ybuf, Ww_bf, W_b, x, out);
}
